// MultiHeadAttention_86887188398792
// MI455X (gfx1250) — compile-verified, matrix operand order checked by reading, not run on hardware
//
#include <hip/hip_runtime.h>
#include <math.h>

#define HIDDEN 1024
#define HEADS 16
#define HEAD_DIM 64
#define BATCH 2
#define SEQ 2048
#define MTOT (BATCH * SEQ) /* 4096 */

typedef __attribute__((ext_vector_type(16))) __bf16 v16bf;
typedef __attribute__((ext_vector_type(8)))  __bf16 v8bf;
typedef __attribute__((ext_vector_type(8)))  float  v8f;

// ---------------------------------------------------------------------------
// 16-bit A/B fragment loader for V_WMMA_F32_16X16X32_BF16 (wave32).
// ISA layout (16-bit A 16x32): lanes 0-15 = rows M=0..15, lanes 16-31 = same
// rows, different K; per lane, elements 0..7 hold K = g*8 + 0..7 and elements
// 8..15 hold K = 16 + g*8 + 0..7 (g = lane/16). B uses the same mapping with
// lane = column N, so we load B from a K-contiguous (transposed) buffer.
// ---------------------------------------------------------------------------
static __device__ inline v16bf load_frag(const __bf16* base, int row,
                                         int row_stride, int k0, int lane) {
  const int g = (lane >> 4) & 1;
  const __bf16* p = base + (size_t)row * (size_t)row_stride + k0;
  union { v16bf v; v8bf h[2]; } u;
  u.h[0] = *(const v8bf*)(p + g * 8);
  u.h[1] = *(const v8bf*)(p + 16 + g * 8);
  return u.v;
}

static __device__ inline v8f wmma_bf16(v16bf a, v16bf b, v8f c) {
  // (neg_a, A, neg_b, B, c_mod, C, reuse_a, reuse_b)
  return __builtin_amdgcn_wmma_f32_16x16x32_bf16(false, a, false, b,
                                                 (short)0, c, false, false);
}

// ---------------------------------------------------------------------------
// Prep: fp32 -> bf16 cast (for x)
// ---------------------------------------------------------------------------
__global__ void cast_f32_bf16_kernel(const float* __restrict__ in,
                                     __bf16* __restrict__ out, int n) {
  int i = blockIdx.x * blockDim.x + threadIdx.x;
  if (i < n) out[i] = (__bf16)in[i];
}

// Prep: fp32 -> bf16 cast + transpose (weights).  wt[n][k] = w[k][n]
__global__ void transpose_cast_kernel(const float* __restrict__ w,
                                      __bf16* __restrict__ wt) {
  int kk = blockIdx.x * 32 + threadIdx.x; // 0..1023 (contiguous in output)
  int nn = blockIdx.y * 8 + threadIdx.y;  // 0..1023
  wt[(size_t)nn * HIDDEN + kk] = (__bf16)w[(size_t)kk * HIDDEN + nn];
}

// ---------------------------------------------------------------------------
// QKV projection: [4096,1024] @ [1024,1024] + bias.
// One wave computes a 32x64 tile: 2 A-fragments x 4 B-fragments -> 8 WMMAs
// per 6 fragment loads (B fragment feeds two consecutive WMMAs).
// blockIdx.z: 0 = Q (scaled by 1/sqrt(D), layout [B,H,S,D] bf16)
//             1 = K (layout [B,H,S,D] bf16)
//             2 = V (layout [B,H,D,S] bf16, transposed for P@V operand)
// ---------------------------------------------------------------------------
__global__ void qkv_gemm_kernel(const __bf16* __restrict__ xb,
                                const __bf16* __restrict__ wqt,
                                const __bf16* __restrict__ wkt,
                                const __bf16* __restrict__ wvt,
                                const float* __restrict__ bq,
                                const float* __restrict__ bk,
                                const float* __restrict__ bv,
                                __bf16* __restrict__ qh,
                                __bf16* __restrict__ kh,
                                __bf16* __restrict__ vth) {
  const int lane = threadIdx.x & 31;
  const int wave = threadIdx.x >> 5;
  const int g = lane >> 4;
  const int ln = lane & 15;
  const int mbase = blockIdx.y * 256 + wave * 32;   // 32 rows per wave
  const int nbase = blockIdx.x * 64;
  const int which = blockIdx.z;

  const __bf16* wt = (which == 0) ? wqt : (which == 1) ? wkt : wvt;
  const float* bias = (which == 0) ? bq : (which == 1) ? bk : bv;

  v8f acc[2][4] = {};
  for (int k0 = 0; k0 < HIDDEN; k0 += 32) {
    v16bf a0 = load_frag(xb, mbase + ln, HIDDEN, k0, lane);
    v16bf a1 = load_frag(xb, mbase + 16 + ln, HIDDEN, k0, lane);
#pragma unroll
    for (int f = 0; f < 4; f++) {
      v16bf b = load_frag(wt, nbase + f * 16 + ln, HIDDEN, k0, lane);
      acc[0][f] = wmma_bf16(a0, b, acc[0][f]);
      acc[1][f] = wmma_bf16(a1, b, acc[1][f]);
    }
  }

#pragma unroll
  for (int t = 0; t < 2; t++) {
#pragma unroll
    for (int f = 0; f < 4; f++) {
      const int n = nbase + f * 16 + ln;    // column index (C layout: lane%16)
      const float bb = bias[n];
      const int h = n >> 6, d = n & 63;
#pragma unroll
      for (int e = 0; e < 8; e++) {
        const int m = mbase + t * 16 + e + g * 8; // C layout: row = e + 8*grp
        const int b2 = m >> 11, s = m & (SEQ - 1);
        float v = acc[t][f][e] + bb;
        if (which == 0) v *= 0.125f;        // fold 1/sqrt(HEAD_DIM) into Q
        if (which == 0)
          qh[(((size_t)(b2 * HEADS + h)) * SEQ + s) * HEAD_DIM + d] = (__bf16)v;
        else if (which == 1)
          kh[(((size_t)(b2 * HEADS + h)) * SEQ + s) * HEAD_DIM + d] = (__bf16)v;
        else
          vth[(((size_t)(b2 * HEADS + h)) * HEAD_DIM + d) * SEQ + s] = (__bf16)v;
      }
    }
  }
}

// ---------------------------------------------------------------------------
// Flash attention: one wave owns 16 query rows; online softmax over 32-key
// chunks; scores and P@V both via v_wmma_f32_16x16x32_bf16.  P is staged
// through per-wave LDS to re-shape C-layout -> A-fragment layout.
// ---------------------------------------------------------------------------
__global__ void flash_attn_kernel(const __bf16* __restrict__ q,
                                  const __bf16* __restrict__ k,
                                  const __bf16* __restrict__ vt,
                                  __bf16* __restrict__ attn) {
  __shared__ __align__(16) __bf16 lds_p[8][16 * 32];

  const int lane = threadIdx.x & 31;
  const int wave = threadIdx.x >> 5;
  const int g = lane >> 4;
  const int ln = lane & 15;
  const int bh = blockIdx.y;                 // 0..31 = b*HEADS + h
  const int qbase = blockIdx.x * 128 + wave * 16;

  const __bf16* qp = q + (size_t)bh * SEQ * HEAD_DIM;
  const __bf16* kp = k + (size_t)bh * SEQ * HEAD_DIM;
  const __bf16* vp = vt + (size_t)bh * HEAD_DIM * SEQ;

  // Q tile fragments: 16 rows x 64 dims -> two 16x32 A fragments
  v16bf aq0 = load_frag(qp, qbase + ln, HEAD_DIM, 0, lane);
  v16bf aq1 = load_frag(qp, qbase + ln, HEAD_DIM, 32, lane);

  v8f o[4] = {};
  float mstate[8], lstate[8];
#pragma unroll
  for (int e = 0; e < 8; e++) { mstate[e] = -1.0e30f; lstate[e] = 0.0f; }

  for (int kc = 0; kc < SEQ; kc += 32) {
    // ---- scores: S = (Q*scale) @ K^T for 32 keys -> two 16x16 f32 tiles
    v8f s0 = {}, s1 = {};
    {
      v16bf b0 = load_frag(kp, kc + ln, HEAD_DIM, 0, lane);
      v16bf b1 = load_frag(kp, kc + ln, HEAD_DIM, 32, lane);
      s0 = wmma_bf16(aq0, b0, s0);
      s0 = wmma_bf16(aq1, b1, s0);
    }
    {
      v16bf b0 = load_frag(kp, kc + 16 + ln, HEAD_DIM, 0, lane);
      v16bf b1 = load_frag(kp, kc + 16 + ln, HEAD_DIM, 32, lane);
      s1 = wmma_bf16(aq0, b0, s1);
      s1 = wmma_bf16(aq1, b1, s1);
    }

    // ---- online softmax, rows live as element e (+8*g), cols across 16 lanes
#pragma unroll
    for (int e = 0; e < 8; e++) {
      float mx = fmaxf(s0[e], s1[e]);
#pragma unroll
      for (int off = 1; off < 16; off <<= 1)
        mx = fmaxf(mx, __shfl_xor(mx, off, 32));
      const float mnew = fmaxf(mstate[e], mx);
      const float scale = __expf(mstate[e] - mnew);
      const float p0 = __expf(s0[e] - mnew);
      const float p1 = __expf(s1[e] - mnew);
      float rs = p0 + p1;
#pragma unroll
      for (int off = 1; off < 16; off <<= 1)
        rs += __shfl_xor(rs, off, 32);
      lstate[e] = lstate[e] * scale + rs;
      mstate[e] = mnew;
      o[0][e] *= scale; o[1][e] *= scale; o[2][e] *= scale; o[3][e] *= scale;

      const int row = e + g * 8;
      lds_p[wave][row * 32 + ln] = (__bf16)p0;
      lds_p[wave][row * 32 + 16 + ln] = (__bf16)p1;
    }
    // LDS is in-order per wave; ensure the DS stores landed before re-reading
    asm volatile("s_wait_dscnt 0" ::: "memory");

    // ---- reload P (16x32 bf16) with the A-fragment mapping
    v16bf ap = load_frag(&lds_p[wave][0], ln, 32, 0, lane);

    // ---- O += P @ V  (V transposed: [d][s], keys contiguous per lane)
#pragma unroll
    for (int f = 0; f < 4; f++) {
      v16bf bvf = load_frag(vp, f * 16 + ln, SEQ, kc, lane);
      o[f] = wmma_bf16(ap, bvf, o[f]);
    }
  }

  // ---- finalize: divide by row sums, scatter to [B*S, HIDDEN] bf16 scratch
  const int b2 = bh >> 4, h = bh & (HEADS - 1);
  float linv[8];
#pragma unroll
  for (int e = 0; e < 8; e++) linv[e] = 1.0f / lstate[e];
#pragma unroll
  for (int f = 0; f < 4; f++) {
    const int d = f * 16 + ln;
#pragma unroll
    for (int e = 0; e < 8; e++) {
      const int s = qbase + e + g * 8;
      attn[((size_t)(b2 * SEQ + s)) * HIDDEN + h * HEAD_DIM + d] =
          (__bf16)(o[f][e] * linv[e]);
    }
  }
}

// ---------------------------------------------------------------------------
// Output projection: out = attn @ Wo + bo, fp32 result.  Same 32x64-per-wave
// register blocking as the QKV GEMM.
// ---------------------------------------------------------------------------
__global__ void oproj_gemm_kernel(const __bf16* __restrict__ attn,
                                  const __bf16* __restrict__ wot,
                                  const float* __restrict__ bo,
                                  float* __restrict__ out) {
  const int lane = threadIdx.x & 31;
  const int wave = threadIdx.x >> 5;
  const int g = lane >> 4;
  const int ln = lane & 15;
  const int mbase = blockIdx.y * 256 + wave * 32;
  const int nbase = blockIdx.x * 64;

  v8f acc[2][4] = {};
  for (int k0 = 0; k0 < HIDDEN; k0 += 32) {
    v16bf a0 = load_frag(attn, mbase + ln, HIDDEN, k0, lane);
    v16bf a1 = load_frag(attn, mbase + 16 + ln, HIDDEN, k0, lane);
#pragma unroll
    for (int f = 0; f < 4; f++) {
      v16bf b = load_frag(wot, nbase + f * 16 + ln, HIDDEN, k0, lane);
      acc[0][f] = wmma_bf16(a0, b, acc[0][f]);
      acc[1][f] = wmma_bf16(a1, b, acc[1][f]);
    }
  }
#pragma unroll
  for (int t = 0; t < 2; t++) {
#pragma unroll
    for (int f = 0; f < 4; f++) {
      const int n = nbase + f * 16 + ln;
      const float bb = bo[n];
#pragma unroll
      for (int e = 0; e < 8; e++) {
        const int m = mbase + t * 16 + e + g * 8;
        out[(size_t)m * HIDDEN + n] = acc[t][f][e] + bb;
      }
    }
  }
}

// ---------------------------------------------------------------------------
extern "C" void kernel_launch(void* const* d_in, const int* in_sizes, int n_in,
                              void* d_out, int out_size, void* d_ws,
                              size_t ws_size, hipStream_t stream) {
  const float* x  = (const float*)d_in[0];
  const float* Wq = (const float*)d_in[1];
  const float* bq = (const float*)d_in[2];
  const float* Wk = (const float*)d_in[3];
  const float* bk = (const float*)d_in[4];
  const float* Wv = (const float*)d_in[5];
  const float* bv = (const float*)d_in[6];
  const float* Wo = (const float*)d_in[7];
  const float* bo = (const float*)d_in[8];
  float* out = (float*)d_out;

  // Workspace carve-up (all bf16): total 48 MB
  char* ws = (char*)d_ws;
  const size_t XB_BYTES = (size_t)MTOT * HIDDEN * 2;          // 8 MB
  const size_t WT_BYTES = (size_t)HIDDEN * HIDDEN * 2;        // 2 MB
  const size_t HD_BYTES = (size_t)BATCH * HEADS * SEQ * HEAD_DIM * 2; // 8 MB
  __bf16* xb  = (__bf16*)(ws);
  __bf16* wqt = (__bf16*)(ws + XB_BYTES);
  __bf16* wkt = (__bf16*)(ws + XB_BYTES + WT_BYTES);
  __bf16* wvt = (__bf16*)(ws + XB_BYTES + 2 * WT_BYTES);
  __bf16* wot = (__bf16*)(ws + XB_BYTES + 3 * WT_BYTES);
  __bf16* qh  = (__bf16*)(ws + XB_BYTES + 4 * WT_BYTES);
  __bf16* kh  = (__bf16*)(ws + XB_BYTES + 4 * WT_BYTES + HD_BYTES);
  __bf16* vth = (__bf16*)(ws + XB_BYTES + 4 * WT_BYTES + 2 * HD_BYTES);
  __bf16* att = (__bf16*)(ws + XB_BYTES + 4 * WT_BYTES + 3 * HD_BYTES);

  // 1) casts / transposes
  {
    const int n = MTOT * HIDDEN;
    cast_f32_bf16_kernel<<<(n + 255) / 256, 256, 0, stream>>>(x, xb, n);
    dim3 blk(32, 8), grd(HIDDEN / 32, HIDDEN / 8);
    transpose_cast_kernel<<<grd, blk, 0, stream>>>(Wq, wqt);
    transpose_cast_kernel<<<grd, blk, 0, stream>>>(Wk, wkt);
    transpose_cast_kernel<<<grd, blk, 0, stream>>>(Wv, wvt);
    transpose_cast_kernel<<<grd, blk, 0, stream>>>(Wo, wot);
  }

  // 2) QKV projections (z = 0/1/2 -> Q/K/V); 256x64 tile per block
  qkv_gemm_kernel<<<dim3(HIDDEN / 64, MTOT / 256, 3), 256, 0, stream>>>(
      xb, wqt, wkt, wvt, bq, bk, bv, qh, kh, vth);

  // 3) attention
  flash_attn_kernel<<<dim3(SEQ / 128, BATCH * HEADS), 256, 0, stream>>>(
      qh, kh, vth, att);

  // 4) output projection
  oproj_gemm_kernel<<<dim3(HIDDEN / 64, MTOT / 256), 256, 0, stream>>>(
      att, wot, bo, out);
}
